// SheafNN_4466765988558
// MI455X (gfx1250) — compile-verified, matrix operand order checked by reading, not run on hardware
//
#include <hip/hip_runtime.h>
#include <hip/hip_bf16.h>
#include <math.h>

#define N_NODES 20000
#define E_HALF  160000
#define E_DIR   320000
#define IN_C    512
#define HID     128
#define OUT_C   40

typedef __bf16 bf16_t;
typedef __attribute__((ext_vector_type(16))) __bf16 v16bf;
typedef __attribute__((ext_vector_type(8)))  float  v8f;
typedef __attribute__((ext_vector_type(4)))  unsigned int u32x4;

// ---------------------------------------------------------------------------
// WMMA helpers (wave32, 16x16x32 bf16 -> f32)
// ---------------------------------------------------------------------------
static __device__ inline v8f wmma_bf16(v16bf a, v16bf b, v8f c) {
  // (neg_a, A, neg_b, B, c_mod, C, reuse_a, reuse_b)
  return __builtin_amdgcn_wmma_f32_16x16x32_bf16(false, a, false, b, (short)0, c,
                                                 false, false);
}

// Load a 16x32 bf16 fragment. Per ISA 7.12.2: lane%16 selects the row,
// lane/16 selects K-half; elements 0..7 cover K = base..base+7 contiguously,
// elements 8..15 cover K = base+16..base+23. Two 16-byte loads per lane.
// Weights stored [out,K] give B[k,col] = W[col,k] -> same indexing (r0=col0).
static __device__ inline v16bf load_frag(const bf16_t* p, int ld, int r0, int k0) {
  int lane = threadIdx.x & 31;
  int r  = r0 + (lane & 15);
  int kh = (lane >> 4) << 3;
  const bf16_t* q = p + (size_t)r * ld + k0 + kh;
  union { v16bf v; u32x4 u[2]; } t;
  t.u[0] = *(const u32x4*)(q);
  t.u[1] = *(const u32x4*)(q + 16);
  return t.v;
}

// A-fragment for the edge MLP: logical row e has data concat(h[src[e]], h[dst[e]])
// of length 2*HID; each aligned 8-element chunk lies fully in one half.
static __device__ inline v16bf load_frag_edge(const bf16_t* hb, const int* src,
                                              const int* dst, int e0, int k0) {
  int lane = threadIdx.x & 31;
  int e  = e0 + (lane & 15);
  int kh = (lane >> 4) << 3;
  int s = src[e], d = dst[e];
  union { v16bf v; u32x4 u[2]; } t;
#pragma unroll
  for (int c = 0; c < 2; ++c) {
    int kk = k0 + kh + c * 16;
    const bf16_t* q = (kk < HID) ? (hb + (size_t)s * HID + kk)
                                 : (hb + (size_t)d * HID + (kk - HID));
    t.u[c] = *(const u32x4*)(q);
  }
  return t.v;
}

// Bias + ReLU a 16x16 f32 tile and park it in LDS as bf16.
static __device__ inline void relu_tile_to_lds(bf16_t* lds, int ldl, int col0,
                                               v8f acc, const float* bias) {
  int lane = threadIdx.x & 31;
  int col = col0 + (lane & 15);
  int rb  = (lane >> 4) << 3;
  float b = bias[col];
#pragma unroll
  for (int r = 0; r < 8; ++r) {
    float v = acc[r] + b;
    lds[(rb + r) * ldl + col] = (bf16_t)(v > 0.f ? v : 0.f);
  }
}

// ---------------------------------------------------------------------------
// Conversion / init kernels
// ---------------------------------------------------------------------------
__global__ void cvt_bf16_kernel(const float* __restrict__ s, bf16_t* __restrict__ d, int n) {
  int i = blockIdx.x * blockDim.x + threadIdx.x;
  if (i < n) d[i] = (bf16_t)s[i];
}

__global__ void padcvt_bf16_kernel(const float* __restrict__ s, bf16_t* __restrict__ d,
                                   int rows_src, int rows_dst, int cols) {
  int i = blockIdx.x * blockDim.x + threadIdx.x;
  if (i >= rows_dst * cols) return;
  int r = i / cols;
  d[i] = (r < rows_src) ? (bf16_t)s[i] : (bf16_t)0.f;
}

__global__ void zero_f32_kernel(float* __restrict__ p, int n) {
  int i = blockIdx.x * blockDim.x + threadIdx.x;
  if (i < n) p[i] = 0.f;
}

// ---------------------------------------------------------------------------
// Node input MLP: h = relu(x @ Win1^T + b1) @ Win2^T + b2     [N,128]
// One block = 16 rows; 4 waves: GEMM1 4 col-tiles (K=512), GEMM2 8 col-tiles.
// ---------------------------------------------------------------------------
__global__ __launch_bounds__(128)
void mlp_in_kernel(const bf16_t* __restrict__ xb, const bf16_t* __restrict__ w1b,
                   const float* __restrict__ b1, const bf16_t* __restrict__ w2b,
                   const float* __restrict__ b2, float* __restrict__ hf,
                   bf16_t* __restrict__ hb) {
  __shared__ __align__(16) bf16_t hs[16 * 64];
  int wave = threadIdx.x >> 5, lane = threadIdx.x & 31;
  int row0 = blockIdx.x * 16;
  int col0 = wave * 16;
  v8f acc = {};
  for (int k0 = 0; k0 < IN_C; k0 += 32) {
    v16bf a = load_frag(xb, IN_C, row0, k0);
    v16bf b = load_frag(w1b, IN_C, col0, k0);
    acc = wmma_bf16(a, b, acc);
  }
  relu_tile_to_lds(hs, 64, col0, acc, b1);
  __syncthreads();
#pragma unroll
  for (int t = 0; t < 2; ++t) {
    int c0 = (wave * 2 + t) * 16;
    v8f a2 = {};
    for (int k0 = 0; k0 < 64; k0 += 32) {
      v16bf a = load_frag(hs, 64, 0, k0);
      v16bf b = load_frag(w2b, 64, c0, k0);
      a2 = wmma_bf16(a, b, a2);
    }
    int col = c0 + (lane & 15);
    int rb  = (lane >> 4) << 3;
    float bias = b2[col];
#pragma unroll
    for (int r = 0; r < 8; ++r) {
      float v = a2[r] + bias;
      size_t idx = (size_t)(row0 + rb + r) * HID + col;
      hf[idx] = v;
      hb[idx] = (bf16_t)v;
    }
  }
}

// ---------------------------------------------------------------------------
// Edge MLP: maps = relu(concat(h[src],h[dst]) @ Wm1^T + b1) @ Wm2^T + b2  [E,4]
// Block = 16 edges; GEMM1 K=256 with fused gather; GEMM2 K=64 on padded Wm2.
// ---------------------------------------------------------------------------
__global__ __launch_bounds__(128)
void edge_mlp_kernel(const bf16_t* __restrict__ hb, const int* __restrict__ src,
                     const int* __restrict__ dst, const bf16_t* __restrict__ w1b,
                     const float* __restrict__ b1, const bf16_t* __restrict__ w2pb,
                     const float* __restrict__ b2, float* __restrict__ maps) {
  __shared__ __align__(16) bf16_t hs[16 * 64];
  int wave = threadIdx.x >> 5, lane = threadIdx.x & 31;
  int e0 = blockIdx.x * 16;
  int col0 = wave * 16;
  v8f acc = {};
  for (int k0 = 0; k0 < 2 * HID; k0 += 32) {
    v16bf a = load_frag_edge(hb, src, dst, e0, k0);
    v16bf b = load_frag(w1b, 2 * HID, col0, k0);
    acc = wmma_bf16(a, b, acc);
  }
  relu_tile_to_lds(hs, 64, col0, acc, b1);
  __syncthreads();
  if (wave == 0) {                 // uniform per-wave branch; EXEC all-1s inside
    v8f a2 = {};
    for (int k0 = 0; k0 < 64; k0 += 32) {
      v16bf a = load_frag(hs, 64, 0, k0);
      v16bf b = load_frag(w2pb, 64, 0, k0);
      a2 = wmma_bf16(a, b, a2);
    }
    int col = lane & 15;
    int rb  = (lane >> 4) << 3;
    if (col < 4) {
      float bias = b2[col];
#pragma unroll
      for (int r = 0; r < 8; ++r)
        maps[(size_t)(e0 + rb + r) * 4 + col] = a2[r] + bias;
    }
  }
}

// ---------------------------------------------------------------------------
// Per-edge: FtF scatter into diag (atomic) and off = -F_src^T F_dst.
// maps[e] is [2,2] row-major: m[j][i] = maps[e*4 + j*2 + i].
// ---------------------------------------------------------------------------
__global__ void edge_maps_kernel(const float* __restrict__ maps, const int* __restrict__ src,
                                 float* __restrict__ diag, float* __restrict__ off) {
  int e = blockIdx.x * blockDim.x + threadIdx.x;
  if (e >= E_DIR) return;
  float m0 = maps[e * 4 + 0], m1 = maps[e * 4 + 1];
  float m2 = maps[e * 4 + 2], m3 = maps[e * 4 + 3];
  float f00 = m0 * m0 + m2 * m2;
  float f01 = m0 * m1 + m2 * m3;
  float f11 = m1 * m1 + m3 * m3;
  int s = src[e];
  atomicAdd(&diag[s * 4 + 0], f00);
  atomicAdd(&diag[s * 4 + 1], f01);
  atomicAdd(&diag[s * 4 + 2], f01);
  atomicAdd(&diag[s * 4 + 3], f11);
  int rev = (e < E_HALF) ? e + E_HALF : e - E_HALF;
  float r0 = maps[rev * 4 + 0], r1 = maps[rev * 4 + 1];
  float r2 = maps[rev * 4 + 2], r3 = maps[rev * 4 + 3];
  off[e * 4 + 0] = -(m0 * r0 + m2 * r2);
  off[e * 4 + 1] = -(m0 * r1 + m2 * r3);
  off[e * 4 + 2] = -(m1 * r0 + m3 * r2);
  off[e * 4 + 3] = -(m1 * r1 + m3 * r3);
}

// ---------------------------------------------------------------------------
// Per-node 2x2 symmetric eigh -> Dinv = V diag(1/sqrt(max(lam,eps))) V^T,
// diag_n = Dinv @ diag @ Dinv.
// ---------------------------------------------------------------------------
__global__ void node_dinv_kernel(const float* __restrict__ diag, float* __restrict__ Dinv,
                                 float* __restrict__ diagn) {
  int n = blockIdx.x * blockDim.x + threadIdx.x;
  if (n >= N_NODES) return;
  float a = diag[n * 4 + 0], b = diag[n * 4 + 1], c = diag[n * 4 + 3];
  float th = 0.5f * atan2f(2.f * b, a - c);
  float ct = cosf(th), st = sinf(th);
  float e1 = a * ct * ct + 2.f * b * ct * st + c * st * st;
  float e2 = a * st * st - 2.f * b * ct * st + c * ct * ct;
  float g1 = rsqrtf(fmaxf(e1, 1e-6f));
  float g2 = rsqrtf(fmaxf(e2, 1e-6f));
  float d00 = g1 * ct * ct + g2 * st * st;
  float d01 = (g1 - g2) * ct * st;
  float d11 = g1 * st * st + g2 * ct * ct;
  Dinv[n * 4 + 0] = d00; Dinv[n * 4 + 1] = d01;
  Dinv[n * 4 + 2] = d01; Dinv[n * 4 + 3] = d11;
  float t00 = d00 * a + d01 * b, t01 = d00 * b + d01 * c;
  float t10 = d01 * a + d11 * b, t11 = d01 * b + d11 * c;
  diagn[n * 4 + 0] = t00 * d00 + t01 * d01;
  diagn[n * 4 + 1] = t00 * d01 + t01 * d11;
  diagn[n * 4 + 2] = t10 * d00 + t11 * d01;
  diagn[n * 4 + 3] = t10 * d01 + t11 * d11;
}

// off_n[e] = Dinv[src] @ off[e] @ Dinv[dst]
__global__ void offn_kernel(const float* __restrict__ off, const float* __restrict__ Dinv,
                            const int* __restrict__ src, const int* __restrict__ dst,
                            float* __restrict__ offn) {
  int e = blockIdx.x * blockDim.x + threadIdx.x;
  if (e >= E_DIR) return;
  const float* ds_ = Dinv + (size_t)src[e] * 4;
  const float* dd  = Dinv + (size_t)dst[e] * 4;
  const float* o   = off + (size_t)e * 4;
  float t00 = ds_[0] * o[0] + ds_[1] * o[2], t01 = ds_[0] * o[1] + ds_[1] * o[3];
  float t10 = ds_[2] * o[0] + ds_[3] * o[2], t11 = ds_[2] * o[1] + ds_[3] * o[3];
  offn[e * 4 + 0] = t00 * dd[0] + t01 * dd[2];
  offn[e * 4 + 1] = t00 * dd[1] + t01 * dd[3];
  offn[e * 4 + 2] = t10 * dd[0] + t11 * dd[2];
  offn[e * 4 + 3] = t10 * dd[1] + t11 * dd[3];
}

// ---------------------------------------------------------------------------
// Diffusion step A: Ht = W1^T (h3 @ W2^T).  h3 viewed as [2N,64] rows.
// Block = 16 nodes = 32 rows; 8 WMMA tiles over 4 waves; W1 mix via LDS.
// ---------------------------------------------------------------------------
__global__ __launch_bounds__(128)
void ht_kernel(const bf16_t* __restrict__ h3b, const bf16_t* __restrict__ w2b,
               const float* __restrict__ w1, float* __restrict__ Ht) {
  __shared__ __align__(16) float xls[32 * 64];
  int wave = threadIdx.x >> 5, lane = threadIdx.x & 31;
  int row0 = blockIdx.x * 32;
#pragma unroll
  for (int t = 0; t < 2; ++t) {
    int idx = wave * 2 + t;
    int rt = idx >> 2, ct = idx & 3;
    v8f acc = {};
    for (int k0 = 0; k0 < 64; k0 += 32) {
      v16bf a = load_frag(h3b, 64, row0 + rt * 16, k0);
      v16bf b = load_frag(w2b, 64, ct * 16, k0);
      acc = wmma_bf16(a, b, acc);
    }
    int col = ct * 16 + (lane & 15);
    int rb  = rt * 16 + ((lane >> 4) << 3);
#pragma unroll
    for (int r = 0; r < 8; ++r) xls[(rb + r) * 64 + col] = acc[r];
  }
  __syncthreads();
  float w00 = w1[0], w01 = w1[1], w10 = w1[2], w11 = w1[3];  // W1[j][i]=w1[j*2+i]
  for (int i = threadIdx.x; i < 16 * HID; i += 128) {
    int nl = i >> 7, rem = i & 127;
    int stlk = rem >> 6, f = rem & 63;
    float x0 = xls[(nl * 2 + 0) * 64 + f];
    float x1 = xls[(nl * 2 + 1) * 64 + f];
    float v = (stlk == 0) ? (w00 * x0 + w10 * x1) : (w01 * x0 + w11 * x1);
    Ht[(size_t)(blockIdx.x * 16 + nl) * HID + rem] = v;
  }
}

// Diffusion step B: msgs = off_n[e] @ Ht[dst], atomically scattered to src.
__global__ void msg_kernel(const float* __restrict__ offn, const float* __restrict__ Ht,
                           const int* __restrict__ src, const int* __restrict__ dst,
                           float* __restrict__ Lacc) {
  int tid = blockIdx.x * blockDim.x + threadIdx.x;
  int e = tid >> 6, f = tid & 63;
  if (e >= E_DIR) return;
  const float* o = offn + (size_t)e * 4;
  int d = dst[e], s = src[e];
  float h0 = Ht[(size_t)d * HID + f], h1 = Ht[(size_t)d * HID + 64 + f];
  atomicAdd(&Lacc[(size_t)s * HID + f],      o[0] * h0 + o[1] * h1);
  atomicAdd(&Lacc[(size_t)s * HID + 64 + f], o[2] * h0 + o[3] * h1);
}

// Diffusion step C: Lh = diag_n @ Ht + Lacc ; h3 -= elu(Lh).
__global__ void update_kernel(const float* __restrict__ diagn, const float* __restrict__ Ht,
                              const float* __restrict__ Lacc, float* __restrict__ hf,
                              bf16_t* __restrict__ hb) {
  int tid = blockIdx.x * blockDim.x + threadIdx.x;
  int n = tid >> 6, f = tid & 63;
  if (n >= N_NODES) return;
  const float* dn = diagn + (size_t)n * 4;
  size_t i0 = (size_t)n * HID + f, i1 = i0 + 64;
  float h0 = Ht[i0], h1 = Ht[i1];
  float l0 = dn[0] * h0 + dn[1] * h1 + Lacc[i0];
  float l1 = dn[2] * h0 + dn[3] * h1 + Lacc[i1];
  float e0 = l0 > 0.f ? l0 : (expf(l0) - 1.f);
  float e1 = l1 > 0.f ? l1 : (expf(l1) - 1.f);
  float v0 = hf[i0] - e0, v1 = hf[i1] - e1;
  hf[i0] = v0; hb[i0] = (bf16_t)v0;
  hf[i1] = v1; hb[i1] = (bf16_t)v1;
}

// ---------------------------------------------------------------------------
// Output MLP: out = relu(h3 @ Wo1^T + b1) @ Wo2^T + b2   [N,40]
// ---------------------------------------------------------------------------
__global__ __launch_bounds__(128)
void mlp_out_kernel(const bf16_t* __restrict__ hb, const bf16_t* __restrict__ w1b,
                    const float* __restrict__ b1, const bf16_t* __restrict__ w2pb,
                    const float* __restrict__ b2, float* __restrict__ out) {
  __shared__ __align__(16) bf16_t hs[16 * 64];
  int wave = threadIdx.x >> 5, lane = threadIdx.x & 31;
  int row0 = blockIdx.x * 16;
  int col0 = wave * 16;
  v8f acc = {};
  for (int k0 = 0; k0 < HID; k0 += 32) {
    v16bf a = load_frag(hb, HID, row0, k0);
    v16bf b = load_frag(w1b, HID, col0, k0);
    acc = wmma_bf16(a, b, acc);
  }
  relu_tile_to_lds(hs, 64, col0, acc, b1);
  __syncthreads();
  if (wave < 3) {                 // 48 padded output columns, 40 valid
    int c0 = wave * 16;
    v8f a2 = {};
    for (int k0 = 0; k0 < 64; k0 += 32) {
      v16bf a = load_frag(hs, 64, 0, k0);
      v16bf b = load_frag(w2pb, 64, c0, k0);
      a2 = wmma_bf16(a, b, a2);
    }
    int col = c0 + (lane & 15);
    int rb  = (lane >> 4) << 3;
    if (col < OUT_C) {
      float bias = b2[col];
#pragma unroll
      for (int r = 0; r < 8; ++r)
        out[(size_t)(row0 + rb + r) * OUT_C + col] = a2[r] + bias;
    }
  }
}

// ---------------------------------------------------------------------------
// Host launcher
// ---------------------------------------------------------------------------
extern "C" void kernel_launch(void* const* d_in, const int* in_sizes, int n_in,
                              void* d_out, int out_size, void* d_ws, size_t ws_size,
                              hipStream_t stream) {
  (void)in_sizes; (void)n_in; (void)out_size; (void)ws_size;
  const float* x      = (const float*)d_in[0];
  const int*   ei     = (const int*)d_in[1];
  const float* w_in1  = (const float*)d_in[2];
  const float* b_in1  = (const float*)d_in[3];
  const float* w_in2  = (const float*)d_in[4];
  const float* b_in2  = (const float*)d_in[5];
  const float* w_map1 = (const float*)d_in[6];
  const float* b_map1 = (const float*)d_in[7];
  const float* w_map2 = (const float*)d_in[8];
  const float* b_map2 = (const float*)d_in[9];
  const float* w_out1 = (const float*)d_in[10];
  const float* b_out1 = (const float*)d_in[11];
  const float* w_out2 = (const float*)d_in[12];
  const float* b_out2 = (const float*)d_in[13];
  const float* W1_0   = (const float*)d_in[14];
  const float* W2_0   = (const float*)d_in[15];
  const float* W1_1   = (const float*)d_in[16];
  const float* W2_1   = (const float*)d_in[17];
  const int* src = ei;
  const int* dst = ei + E_DIR;

  // bump allocator over workspace (~73 MB total)
  char* ws = (char*)d_ws;
  size_t ofs = 0;
  auto alloc = [&](size_t bytes) -> void* {
    void* p = ws + ofs;
    ofs = (ofs + bytes + 255) & ~(size_t)255;
    return p;
  };
  bf16_t* xb       = (bf16_t*)alloc((size_t)N_NODES * IN_C * 2);
  bf16_t* w_in1b   = (bf16_t*)alloc(64 * IN_C * 2);
  bf16_t* w_in2b   = (bf16_t*)alloc(HID * 64 * 2);
  bf16_t* w_map1b  = (bf16_t*)alloc(64 * 2 * HID * 2);
  bf16_t* w_map2pb = (bf16_t*)alloc(16 * 64 * 2);
  bf16_t* w_out1b  = (bf16_t*)alloc(64 * HID * 2);
  bf16_t* w_out2pb = (bf16_t*)alloc(48 * 64 * 2);
  bf16_t* W2_0b    = (bf16_t*)alloc(64 * 64 * 2);
  bf16_t* W2_1b    = (bf16_t*)alloc(64 * 64 * 2);
  bf16_t* hb       = (bf16_t*)alloc((size_t)N_NODES * HID * 2);
  float*  hf       = (float*)alloc((size_t)N_NODES * HID * 4);
  float*  maps     = (float*)alloc((size_t)E_DIR * 4 * 4);
  float*  offm     = (float*)alloc((size_t)E_DIR * 4 * 4);
  float*  offn     = (float*)alloc((size_t)E_DIR * 4 * 4);
  float*  diag     = (float*)alloc((size_t)N_NODES * 4 * 4);
  float*  Dinv     = (float*)alloc((size_t)N_NODES * 4 * 4);
  float*  diagn    = (float*)alloc((size_t)N_NODES * 4 * 4);
  float*  Ht       = (float*)alloc((size_t)N_NODES * HID * 4);
  float*  Lacc     = (float*)alloc((size_t)N_NODES * HID * 4);

  const int T = 256;
  auto g = [](int n, int t) { return (n + t - 1) / t; };

  cvt_bf16_kernel<<<g(N_NODES * IN_C, T), T, 0, stream>>>(x, xb, N_NODES * IN_C);
  cvt_bf16_kernel<<<g(64 * IN_C, T), T, 0, stream>>>(w_in1, w_in1b, 64 * IN_C);
  cvt_bf16_kernel<<<g(HID * 64, T), T, 0, stream>>>(w_in2, w_in2b, HID * 64);
  cvt_bf16_kernel<<<g(64 * 2 * HID, T), T, 0, stream>>>(w_map1, w_map1b, 64 * 2 * HID);
  cvt_bf16_kernel<<<g(64 * HID, T), T, 0, stream>>>(w_out1, w_out1b, 64 * HID);
  cvt_bf16_kernel<<<g(64 * 64, T), T, 0, stream>>>(W2_0, W2_0b, 64 * 64);
  cvt_bf16_kernel<<<g(64 * 64, T), T, 0, stream>>>(W2_1, W2_1b, 64 * 64);
  padcvt_bf16_kernel<<<g(16 * 64, T), T, 0, stream>>>(w_map2, w_map2pb, 4, 16, 64);
  padcvt_bf16_kernel<<<g(48 * 64, T), T, 0, stream>>>(w_out2, w_out2pb, OUT_C, 48, 64);

  mlp_in_kernel<<<N_NODES / 16, 128, 0, stream>>>(xb, w_in1b, b_in1, w_in2b, b_in2, hf, hb);
  edge_mlp_kernel<<<E_DIR / 16, 128, 0, stream>>>(hb, src, dst, w_map1b, b_map1,
                                                  w_map2pb, b_map2, maps);
  zero_f32_kernel<<<g(N_NODES * 4, T), T, 0, stream>>>(diag, N_NODES * 4);
  edge_maps_kernel<<<g(E_DIR, T), T, 0, stream>>>(maps, src, diag, offm);
  node_dinv_kernel<<<g(N_NODES, T), T, 0, stream>>>(diag, Dinv, diagn);
  offn_kernel<<<g(E_DIR, T), T, 0, stream>>>(offm, Dinv, src, dst, offn);

  for (int layer = 0; layer < 2; ++layer) {
    const float*  w1  = (layer == 0) ? W1_0  : W1_1;
    const bf16_t* w2b = (layer == 0) ? W2_0b : W2_1b;
    ht_kernel<<<N_NODES / 16, 128, 0, stream>>>(hb, w2b, w1, Ht);
    zero_f32_kernel<<<g(N_NODES * HID, T), T, 0, stream>>>(Lacc, N_NODES * HID);
    msg_kernel<<<g(E_DIR * 64, T), T, 0, stream>>>(offn, Ht, src, dst, Lacc);
    update_kernel<<<g(N_NODES * 64, T), T, 0, stream>>>(diagn, Ht, Lacc, hf, hb);
  }

  mlp_out_kernel<<<N_NODES / 16, 128, 0, stream>>>(hb, w_out1b, b_out1, w_out2pb, b_out2,
                                                   (float*)d_out);
}
